// GPT2Block_4337916970022
// MI455X (gfx1250) — matrix operand order repaired and checked by reading, not compiled, not run on hardware
//
#include <hip/hip_runtime.h>
#include <stdint.h>

// ---------------------------------------------------------------------------
// GPT-2 block for MI455X (gfx1250, wave32, WMMA 16x16x32 bf16, TDM tile DMA)
// ---------------------------------------------------------------------------

typedef __bf16 bf16;
typedef bf16  bf16x16 __attribute__((ext_vector_type(16)));
typedef float v8f     __attribute__((ext_vector_type(8)));

typedef unsigned int tdm_u32x4 __attribute__((ext_vector_type(4)));
typedef int          tdm_i32x4 __attribute__((ext_vector_type(4)));
typedef int          tdm_i32x8 __attribute__((ext_vector_type(8)));

union Frag {
  bf16x16 v;
  uint4   u[2];
};

constexpr int S_LEN = 2048;
constexpr int NB    = 2;
constexpr int D_EMB = 1024;
constexpr int H_NUM = 16;
constexpr int HD    = 64;
constexpr int D_FF  = 4096;
constexpr int M_TOK = NB * S_LEN;          // 4096 token rows

constexpr int EPI_QKV = 0, EPI_RES = 1, EPI_GELU = 2;

#if defined(__has_builtin)
#if __has_builtin(__builtin_amdgcn_tensor_load_to_lds)
#define USE_TDM 1
#if __has_include(<hip/amd_detail/amd_gfx1250_TDM.h>)
#define TDM_6ARG 1
#else
#define TDM_6ARG 0
#endif
#endif
#endif
#ifndef USE_TDM
#define USE_TDM 0
#endif

__device__ __forceinline__ v8f wmma_bf16(const Frag& a, const Frag& b, v8f c) {
  return __builtin_amdgcn_wmma_f32_16x16x32_bf16(
      /*neg_a=*/false, a.v, /*neg_b=*/false, b.v,
      /*c_mod=*/(short)0, c, /*reuse_a=*/false, /*reuse_b=*/false);
}

#if USE_TDM
// DMA a 128-row x 32-element bf16 tile (row stride K elements in memory) into
// LDS at byte offset `lds_off`, inserting 16B of padding after every 64B row
// so the LDS layout matches bf16[128][40].  One issue per wave; TENSORcnt.
__device__ __forceinline__ void tdm_tile_load(uint32_t lds_off,
                                              const bf16* gptr, uint32_t K) {
  const uint64_t ga = (uint64_t)(uintptr_t)gptr;
  tdm_u32x4 G0 = {1u,                       // count=1 (valid descriptor)
                  lds_off,                  // lds_addr
                  (uint32_t)ga,             // global_addr[31:0]
                  (uint32_t)((ga >> 32) & 0x1FFFFFFu) | (2u << 30)};  // type=2
  tdm_i32x8 G1 = {0, 0, 0, 0, 0, 0, 0, 0};
  // data_size=2B (code 1), pad_enable, pad_interval: 16 DWORDs (code 3),
  // pad_amount: 4 DWORDs (code 3)
  G1[0] = (int)((1u << 16) | (1u << 20) | (3u << 22) | (3u << 25));
  G1[1] = (int)((K & 0xFFFFu) << 16);                    // tensor_dim0[15:0]
  G1[2] = (int)(((K >> 16) & 0xFFFFu) | (128u << 16));   // dim0 hi | dim1=128
  G1[3] = (int)(32u << 16);                              // tile_dim0 = 32
  G1[4] = (int)128u;                                     // tile_dim1 = 128
  G1[5] = (int)K;                                        // dim0_stride[31:0]
  tdm_i32x4 Z4 = {0, 0, 0, 0};
#if TDM_6ARG
  tdm_i32x8 Z8 = {0, 0, 0, 0, 0, 0, 0, 0};
  __builtin_amdgcn_tensor_load_to_lds(G0, G1, Z4, Z4, Z8, 0);
#else
  __builtin_amdgcn_tensor_load_to_lds(G0, G1, Z4, Z4, 0);
#endif
}
#endif

// ---------------------------------------------------------------------------
// Weight convert + transpose:  W[K,N] f32  ->  Wt[N,K] bf16
// ---------------------------------------------------------------------------
__global__ __launch_bounds__(256) void wt_cvt_kernel(const float* __restrict__ W,
                                                     bf16* __restrict__ Wt,
                                                     int K, int N) {
  size_t idx = (size_t)blockIdx.x * 256 + threadIdx.x;
  if (idx >= (size_t)K * N) return;
  int n = (int)(idx / K);
  int k = (int)(idx % K);
  Wt[idx] = (bf16)W[(size_t)k * N + n];
}

// ---------------------------------------------------------------------------
// LayerNorm over D=1024, output bf16.  One block (256 thr) per row.
// ---------------------------------------------------------------------------
__global__ __launch_bounds__(256) void ln_bf16_kernel(const float* __restrict__ x,
                                                      const float* __restrict__ w,
                                                      const float* __restrict__ b,
                                                      bf16* __restrict__ out) {
  __shared__ float rs[256], rq[256];
  const int row = blockIdx.x;
  const int tid = threadIdx.x;
  const float4 v = ((const float4*)(x + (size_t)row * D_EMB))[tid];
  rs[tid] = v.x + v.y + v.z + v.w;
  rq[tid] = v.x * v.x + v.y * v.y + v.z * v.z + v.w * v.w;
  __syncthreads();
  for (int o = 128; o > 0; o >>= 1) {
    if (tid < o) { rs[tid] += rs[tid + o]; rq[tid] += rq[tid + o]; }
    __syncthreads();
  }
  const float mu   = rs[0] * (1.f / D_EMB);
  const float var  = rq[0] * (1.f / D_EMB) - mu * mu;
  const float rstd = rsqrtf(var + 1e-5f);
  const float vv[4] = {v.x, v.y, v.z, v.w};
  bf16* o4 = out + (size_t)row * D_EMB + tid * 4;
#pragma unroll
  for (int j = 0; j < 4; ++j) {
    const int c = tid * 4 + j;
    o4[j] = (bf16)((vv[j] - mu) * rstd * w[c] + b[c]);
  }
}

// ---------------------------------------------------------------------------
// Tiled bf16 WMMA GEMM:  C[M,N] = A[M,K] * Bt[N,K]^T  (+ epilogue)
// 256 threads = 8 waves; block tile 128x128; K-step 32; wave tile 64x32.
// Tiles staged in LDS by the Tensor Data Mover (double buffered) when
// available, else by cooperative b128 loads.
// ---------------------------------------------------------------------------
template <int EPI>
__global__ __launch_bounds__(256) void gemm_bf16_kernel(
    const bf16* __restrict__ A, const bf16* __restrict__ Bt,
    const float* __restrict__ bias, const float* __restrict__ res,
    void* __restrict__ out, int N, int K) {
  __shared__ bf16 Al[2][128][40];   // 32 valid + 8 pad per row
  __shared__ bf16 Bl[2][128][40];

  const int tid  = threadIdx.x;
  const int lane = tid & 31, wv = tid >> 5;
  const int wm = wv >> 2, wn = wv & 3;          // 2 x 4 wave grid
  const int blockM = blockIdx.y * 128;
  const int blockN = blockIdx.x * 128;
  const int lrow = lane & 15;
  const int ahk  = (lane >> 4) * 8;             // A-frag K offset (elements)
  const int bhk  = (lane >> 4) * 8;             // B-frag K offset (elements)

  v8f acc[4][2] = {};

#if USE_TDM
  const uint32_t ldsA[2] = {(uint32_t)(uintptr_t)&Al[0][0][0],
                            (uint32_t)(uintptr_t)&Al[1][0][0]};
  const uint32_t ldsB[2] = {(uint32_t)(uintptr_t)&Bl[0][0][0],
                            (uint32_t)(uintptr_t)&Bl[1][0][0]};
  if (wv == 0) {
    tdm_tile_load(ldsA[0], A + (size_t)blockM * K, (uint32_t)K);
    tdm_tile_load(ldsB[0], Bt + (size_t)blockN * K, (uint32_t)K);
    __builtin_amdgcn_s_wait_tensorcnt(0);
  }
  __syncthreads();
#endif

  for (int kt = 0; kt < K; kt += 32) {
#if USE_TDM
    const int cur = (kt >> 5) & 1;
    if (wv == 0 && kt + 32 < K) {               // prefetch next tile via TDM
      tdm_tile_load(ldsA[cur ^ 1], A + (size_t)blockM * K + kt + 32,
                    (uint32_t)K);
      tdm_tile_load(ldsB[cur ^ 1], Bt + (size_t)blockN * K + kt + 32,
                    (uint32_t)K);
    }
#else
    const int cur = 0;
    __syncthreads();
#pragma unroll
    for (int i = 0; i < 2; ++i) {               // 512 x 16B chunks / 256 thr
      const int c = tid + i * 256;
      const int row = c >> 2, off = (c & 3) * 8;
      *(uint4*)&Al[0][row][off] =
          *(const uint4*)(A + (size_t)(blockM + row) * K + kt + off);
      *(uint4*)&Bl[0][row][off] =
          *(const uint4*)(Bt + (size_t)(blockN + row) * K + kt + off);
    }
    __syncthreads();
#endif

    Frag af[4], bfr[2];
#pragma unroll
    for (int tm = 0; tm < 4; ++tm) {
      const bf16* p = &Al[cur][wm * 64 + tm * 16 + lrow][ahk];
      af[tm].u[0] = *(const uint4*)p;
      af[tm].u[1] = *(const uint4*)(p + 16);
    }
#pragma unroll
    for (int tn = 0; tn < 2; ++tn) {
      const bf16* p = &Bl[cur][wn * 32 + tn * 16 + lrow][bhk];
      bfr[tn].u[0] = *(const uint4*)p;
      bfr[tn].u[1] = *(const uint4*)(p + 16);
    }
#pragma unroll
    for (int tm = 0; tm < 4; ++tm)
#pragma unroll
      for (int tn = 0; tn < 2; ++tn)
        acc[tm][tn] = wmma_bf16(af[tm], bfr[tn], acc[tm][tn]);

#if USE_TDM
    if (wv == 0) __builtin_amdgcn_s_wait_tensorcnt(0);
    __syncthreads();
#endif
  }

  // ---- epilogue: C layout (lane = column, vgpr r = row r + 8*(lane>=16)) ----
  const int m0 = blockM + wm * 64;
  const int n0 = blockN + wn * 32;
  const int mh = (lane >> 4) * 8;
#pragma unroll
  for (int tm = 0; tm < 4; ++tm) {
#pragma unroll
    for (int tn = 0; tn < 2; ++tn) {
#pragma unroll
      for (int r = 0; r < 8; ++r) {
        const int m = m0 + tm * 16 + r + mh;
        const int n = n0 + tn * 16 + lrow;
        float val = acc[tm][tn][r] + bias[n];
        if constexpr (EPI == EPI_QKV) {
          bf16* qkv = (bf16*)out;
          const int bq = m >> 11, sq = m & (S_LEN - 1);
          if (n < D_EMB) {                      // Q, head-major, pre-scaled
            const int hh = n >> 6, dd = n & 63;
            qkv[((size_t)(bq * H_NUM + hh) * S_LEN + sq) * HD + dd] =
                (bf16)(val * 0.125f);
          } else if (n < 2 * D_EMB) {           // K, head-major
            const int nn = n - D_EMB, hh = nn >> 6, dd = nn & 63;
            qkv[(size_t)M_TOK * D_EMB +
                ((size_t)(bq * H_NUM + hh) * S_LEN + sq) * HD + dd] = (bf16)val;
          } else {                              // V, transposed [B,H,64,S]
            const int nn = n - 2 * D_EMB, hh = nn >> 6, dd = nn & 63;
            qkv[(size_t)2 * M_TOK * D_EMB +
                ((size_t)(bq * H_NUM + hh) * HD + dd) * S_LEN + sq] = (bf16)val;
          }
        } else if constexpr (EPI == EPI_RES) {
          float* o = (float*)out;
          o[(size_t)m * N + n] = val + res[(size_t)m * N + n];
        } else {                                // tanh-GELU via sigmoid form
          const float x = val;
          const float u = x + 0.044715f * x * x * x;
          const float g = x / (1.f + __expf(-1.5957691216057308f * u));
          ((bf16*)out)[(size_t)m * N + n] = (bf16)g;
        }
      }
    }
  }
}

// ---------------------------------------------------------------------------
// Flash-style causal attention. Grid (S/64, H, B), 128 threads = 4 waves,
// each wave owns 16 queries; streams 32-key tiles; Q pre-scaled by 1/sqrt(hd).
// ---------------------------------------------------------------------------
__global__ __launch_bounds__(128) void attn_kernel(
    const bf16* __restrict__ Qg, const bf16* __restrict__ Kg,
    const bf16* __restrict__ Vt, bf16* __restrict__ ctx) {
  __shared__ bf16 Pl[4][16][40];                // per-wave P scratch

  const int lane = threadIdx.x & 31, wv = threadIdx.x >> 5;
  const int h = blockIdx.y, bq = blockIdx.z;
  const int qbase = blockIdx.x * 64 + wv * 16;
  const int lrow = lane & 15;
  const int half = lane >> 4;
  const size_t headQK = (size_t)(bq * H_NUM + h) * S_LEN * HD;  // Q/K base
  const size_t headV  = (size_t)(bq * H_NUM + h) * HD * S_LEN;  // Vt base

  Frag qf[2];
  {
    const bf16* qp = Qg + headQK + (size_t)(qbase + lrow) * HD + half * 8;
    qf[0].u[0] = *(const uint4*)qp;
    qf[0].u[1] = *(const uint4*)(qp + 16);
    qf[1].u[0] = *(const uint4*)(qp + 32);
    qf[1].u[1] = *(const uint4*)(qp + 48);
  }

  float m_i[8], l_i[8];
  v8f o[4] = {};
#pragma unroll
  for (int r = 0; r < 8; ++r) { m_i[r] = -1e30f; l_i[r] = 0.f; }

  for (int kb = 0; kb <= qbase + 15; kb += 32) {
    // scores = Q(16x64) x K^T(64x32): B-frag lane = key column, K = hd range
    Frag kf[2][2];
#pragma unroll
    for (int nt = 0; nt < 2; ++nt) {
      const bf16* kp =
          Kg + headQK + (size_t)(kb + nt * 16 + lrow) * HD + half * 8;
#pragma unroll
      for (int kh = 0; kh < 2; ++kh) {
        kf[nt][kh].u[0] = *(const uint4*)(kp + kh * 32);
        kf[nt][kh].u[1] = *(const uint4*)(kp + kh * 32 + 16);
      }
    }
    v8f s0 = {}, s1 = {};
    s0 = wmma_bf16(qf[0], kf[0][0], s0);
    s0 = wmma_bf16(qf[1], kf[0][1], s0);
    s1 = wmma_bf16(qf[0], kf[1][0], s1);
    s1 = wmma_bf16(qf[1], kf[1][1], s1);

    float p0[8], p1[8], scal[8];
#pragma unroll
    for (int r = 0; r < 8; ++r) {
      const int qi = qbase + r + half * 8;
      float a = s0[r], b = s1[r];
      if (kb + lrow > qi)      a = -1e30f;      // causal mask
      if (kb + 16 + lrow > qi) b = -1e30f;
      float mx = fmaxf(a, b);
#pragma unroll
      for (int off = 1; off < 16; off <<= 1)
        mx = fmaxf(mx, __shfl_xor(mx, off, 32));
      const float mn = fmaxf(m_i[r], mx);
      const float sc = __expf(m_i[r] - mn);
      const float e0 = __expf(a - mn);
      const float e1 = __expf(b - mn);
      float rsum = e0 + e1;
#pragma unroll
      for (int off = 1; off < 16; off <<= 1)
        rsum += __shfl_xor(rsum, off, 32);
      l_i[r] = l_i[r] * sc + rsum;
      m_i[r] = mn;
      p0[r] = e0; p1[r] = e1; scal[r] = sc;
    }
#pragma unroll
    for (int nt = 0; nt < 4; ++nt)
#pragma unroll
      for (int r = 0; r < 8; ++r) o[nt][r] *= scal[r];

    // P (C-layout f32) -> LDS [m][k] bf16 -> A-frag
#pragma unroll
    for (int r = 0; r < 8; ++r) {
      Pl[wv][r + half * 8][lrow]      = (bf16)p0[r];
      Pl[wv][r + half * 8][lrow + 16] = (bf16)p1[r];
    }
    asm volatile("s_wait_dscnt 0" ::: "memory");
    Frag pf;
    {
      const bf16* pp = &Pl[wv][lrow][half * 8];
      pf.u[0] = *(const uint4*)pp;
      pf.u[1] = *(const uint4*)(pp + 16);
    }
    // ctx += P(16x32) x V(32x64): Vt rows are key-contiguous per hd column
#pragma unroll
    for (int nt = 0; nt < 4; ++nt) {
      Frag vf;
      const bf16* vp =
          Vt + headV + (size_t)(nt * 16 + lrow) * S_LEN + kb + half * 8;
      vf.u[0] = *(const uint4*)vp;
      vf.u[1] = *(const uint4*)(vp + 16);
      o[nt] = wmma_bf16(pf, vf, o[nt]);
    }
  }

#pragma unroll
  for (int r = 0; r < 8; ++r) {
    const float inv = (l_i[r] > 0.f) ? 1.f / l_i[r] : 0.f;
    const int sq = qbase + r + half * 8;
    const size_t rowo = ((size_t)bq * S_LEN + sq) * D_EMB + h * HD;
#pragma unroll
    for (int nt = 0; nt < 4; ++nt)
      ctx[rowo + nt * 16 + lrow] = (bf16)(o[nt][r] * inv);
  }
}

// ---------------------------------------------------------------------------
extern "C" void kernel_launch(void* const* d_in, const int* in_sizes, int n_in,
                              void* d_out, int out_size, void* d_ws,
                              size_t ws_size, hipStream_t stream) {
  const float* hs   = (const float*)d_in[0];
  const float* ln1w = (const float*)d_in[1];
  const float* ln1b = (const float*)d_in[2];
  const float* ln2w = (const float*)d_in[3];
  const float* ln2b = (const float*)d_in[4];
  const float* Wqkv = (const float*)d_in[5];
  const float* bqkv = (const float*)d_in[6];
  const float* Wpa  = (const float*)d_in[7];
  const float* bpa  = (const float*)d_in[8];
  const float* Wfc  = (const float*)d_in[9];
  const float* bfc  = (const float*)d_in[10];
  const float* Wpm  = (const float*)d_in[11];
  const float* bpm  = (const float*)d_in[12];

  char* p = (char*)d_ws;
  auto alloc = [&](size_t bytes) {
    char* r = p;
    p += (bytes + 255) & ~(size_t)255;
    return r;
  };
  bf16*  X1   = (bf16*)alloc((size_t)M_TOK * D_EMB * 2);
  bf16*  Wq_t = (bf16*)alloc((size_t)3 * D_EMB * D_EMB * 2);
  bf16*  Wp_t = (bf16*)alloc((size_t)D_EMB * D_EMB * 2);
  bf16*  Wf_t = (bf16*)alloc((size_t)D_FF * D_EMB * 2);
  bf16*  Wm_t = (bf16*)alloc((size_t)D_EMB * D_FF * 2);
  bf16*  QKV  = (bf16*)alloc((size_t)3 * M_TOK * D_EMB * 2);
  bf16*  CTX  = (bf16*)alloc((size_t)M_TOK * D_EMB * 2);
  float* HO   = (float*)alloc((size_t)M_TOK * D_EMB * 4);
  bf16*  X2   = (bf16*)alloc((size_t)M_TOK * D_EMB * 2);
  bf16*  H2   = (bf16*)alloc((size_t)M_TOK * D_FF * 2);

  // Per-call weight convert+transpose (deterministic, graph-capture safe)
  wt_cvt_kernel<<<(3 * D_EMB * D_EMB + 255) / 256, 256, 0, stream>>>(
      Wqkv, Wq_t, D_EMB, 3 * D_EMB);
  wt_cvt_kernel<<<(D_EMB * D_EMB + 255) / 256, 256, 0, stream>>>(
      Wpa, Wp_t, D_EMB, D_EMB);
  wt_cvt_kernel<<<(D_EMB * D_FF + 255) / 256, 256, 0, stream>>>(
      Wfc, Wf_t, D_EMB, D_FF);
  wt_cvt_kernel<<<(D_FF * D_EMB + 255) / 256, 256, 0, stream>>>(
      Wpm, Wm_t, D_FF, D_EMB);

  // LN1 -> bf16
  ln_bf16_kernel<<<M_TOK, 256, 0, stream>>>(hs, ln1w, ln1b, X1);

  // QKV GEMM (scatter to Q head-major / K head-major / V transposed)
  gemm_bf16_kernel<EPI_QKV>
      <<<dim3(3 * D_EMB / 128, M_TOK / 128), 256, 0, stream>>>(
          X1, Wq_t, bqkv, nullptr, QKV, 3 * D_EMB, D_EMB);

  // Causal flash attention
  attn_kernel<<<dim3(S_LEN / 64, H_NUM, NB), 128, 0, stream>>>(
      QKV, QKV + (size_t)M_TOK * D_EMB, QKV + (size_t)2 * M_TOK * D_EMB, CTX);

  // attn out-proj + residual -> f32
  gemm_bf16_kernel<EPI_RES>
      <<<dim3(D_EMB / 128, M_TOK / 128), 256, 0, stream>>>(
          CTX, Wp_t, bpa, hs, HO, D_EMB, D_EMB);

  // LN2 -> bf16
  ln_bf16_kernel<<<M_TOK, 256, 0, stream>>>(HO, ln2w, ln2b, X2);

  // FC + GELU -> bf16
  gemm_bf16_kernel<EPI_GELU>
      <<<dim3(D_FF / 128, M_TOK / 128), 256, 0, stream>>>(
          X2, Wf_t, bfc, nullptr, H2, D_FF, D_EMB);

  // MLP proj + residual -> f32 output
  gemm_bf16_kernel<EPI_RES>
      <<<dim3(D_EMB / 128, M_TOK / 128), 256, 0, stream>>>(
          H2, Wm_t, bpm, HO, d_out, D_EMB, D_FF);
}
